// TransformerEncoderLayer_57509612094059
// MI455X (gfx1250) — matrix operand order repaired and checked by reading, not compiled, not run on hardware
//
#include <hip/hip_runtime.h>

// ---------------------------------------------------------------------------
// TransformerEncoderLayer for MI455X (gfx1250, wave32, WMMA 16x16x32 f16,
// async global->LDS staging for the attention V slab).
// B=4 T=1024 D=256 H=8 DK=DH=32 CS=3 PH=16
// ---------------------------------------------------------------------------

typedef _Float16 half_t;
typedef __attribute__((ext_vector_type(16))) _Float16 v16h;
typedef __attribute__((ext_vector_type(8)))  float    v8f;

#define B_   4
#define T_   1024
#define D_   256
#define H_   8
#define DK_  32
#define DH_  32
#define CS_  3
#define PH_  16
#define BT_  (B_ * T_)

__device__ __constant__ float kNegBig = -1000000000.0f;
__device__ __constant__ float kNegThr = -10000000.0f;
#define LN_EPS       1e-5f
#define INV_SQRT_DK  0.17677669529663687f   // 1/sqrt(32)

// ---------------------------------------------------------------------------
// Fragment loaders (16-bit, wave32). Sources are "fragment-major x K" row-major.
// A 16x32 layout (ISA 7.12.2): lane half 0 -> K {0..7,16..23}, half 1 -> +8.
// B 32x16 layout: same K order as A (half 0 -> K {0..7,16..23}, half 1 -> +8).
// ---------------------------------------------------------------------------
__device__ __forceinline__ v16h load_a16(const half_t* base, int ld, int lane) {
  const half_t* p = base + (lane & 15) * ld + ((lane >> 4) << 3);
  v16h f;
#pragma unroll
  for (int j = 0; j < 4; ++j) {
    f[2 * j]     = p[2 * j];
    f[2 * j + 1] = p[2 * j + 1];
    f[8 + 2 * j]     = p[16 + 2 * j];
    f[8 + 2 * j + 1] = p[16 + 2 * j + 1];
  }
  return f;
}

__device__ __forceinline__ v16h load_b16(const half_t* base, int ld, int lane) {
  const half_t* p = base + (lane & 15) * ld + ((lane >> 4) << 3);
  v16h f;
#pragma unroll
  for (int e = 0; e < 16; ++e) f[e] = p[e < 8 ? e : e + 8];
  return f;
}

__device__ __forceinline__ v8f wmma16(v16h a, v16h b, v8f c) {
  return __builtin_amdgcn_wmma_f32_16x16x32_f16(false, a, false, b,
                                                (short)0, c, false, false);
}

// Async global->LDS 16-byte copy (ASYNCcnt-tracked, no VGPR staging).
__device__ __forceinline__ void async_ld_b128(unsigned lds_byte_off,
                                              const half_t* gptr) {
  asm volatile("global_load_async_to_lds_b128 %0, %1, off"
               :: "v"(lds_byte_off), "v"(gptr) : "memory");
}
__device__ __forceinline__ void wait_async0() {
  asm volatile("s_wait_asynccnt 0x0" ::: "memory");
}

// ---------------------------------------------------------------------------
// Prep kernels: cast x, transpose+cast weights into fragment-major layouts.
// ---------------------------------------------------------------------------
__global__ void tel_cast_x(const float* __restrict__ x, half_t* __restrict__ xb,
                           int n) {
  int i = blockIdx.x * blockDim.x + threadIdx.x;
  if (i < n) xb[i] = (half_t)x[i];
}

__global__ void tel_prep_w(const float* __restrict__ Wq, const float* __restrict__ Wk,
                           const float* __restrict__ Wv, const float* __restrict__ Wo,
                           const float* __restrict__ Wc,
                           half_t* __restrict__ Wqt, half_t* __restrict__ Wkt,
                           half_t* __restrict__ Wvt, half_t* __restrict__ Wot,
                           half_t* __restrict__ Wcb) {
  const int i = blockIdx.x * blockDim.x + threadIdx.x;   // 0 .. 65535
  const int nrow = i >> 8;        // output-column index (fragment-major row)
  const int d    = i & 255;       // K index
  const int h = nrow >> 5, k = nrow & 31;
  Wqt[i] = (half_t)Wq[(h * D_ + d) * DK_ + k];
  Wkt[i] = (half_t)Wk[(h * D_ + d) * DK_ + k];
  Wvt[i] = (half_t)Wv[(h * D_ + d) * DH_ + k];
  Wot[i] = (half_t)Wo[d * D_ + nrow];
#pragma unroll
  for (int c = 0; c < CS_; ++c)
    Wcb[c * D_ * D_ + i] = (half_t)Wc[(nrow * D_ + d) * CS_ + c];
}

// ---------------------------------------------------------------------------
// QKV projection: q/k stored [B,H,T,32], v stored transposed [B,H,32,T].
// One wave per 16x32 output tile (A fragment reused for two B tiles).
// ---------------------------------------------------------------------------
__global__ __launch_bounds__(32) void tel_qkv_gemm(
    const half_t* __restrict__ xb,
    const half_t* __restrict__ Wqt, const half_t* __restrict__ Wkt,
    const half_t* __restrict__ Wvt,
    half_t* __restrict__ qb, half_t* __restrict__ kb, half_t* __restrict__ vtb) {
  const int lane = threadIdx.x;
  const int m0 = blockIdx.x << 4;
  const int n0 = blockIdx.y << 5;          // two 16-wide n-tiles per wave
  const int which = blockIdx.z;
  const half_t* Wt = (which == 0) ? Wqt : (which == 1) ? Wkt : Wvt;
  v8f acc0 = {}, acc1 = {};
  for (int kc = 0; kc < D_; kc += 32) {
    if (kc + 32 < D_) __builtin_prefetch(xb + m0 * D_ + kc + 32, 0, 1);
    v16h a  = load_a16(xb + m0 * D_ + kc, D_, lane);
    v16h b0 = load_b16(Wt + n0 * D_ + kc, D_, lane);
    v16h b1 = load_b16(Wt + (n0 + 16) * D_ + kc, D_, lane);
    acc0 = wmma16(a, b0, acc0);
    acc1 = wmma16(a, b1, acc1);
  }
  const int half_ = lane >> 4;
#pragma unroll
  for (int nt = 0; nt < 2; ++nt) {
    const v8f& acc = nt ? acc1 : acc0;
    const int gcol = n0 + nt * 16 + (lane & 15);
    const int h = gcol >> 5, sub = gcol & 31;
#pragma unroll
    for (int r = 0; r < 8; ++r) {
      const int grow = m0 + r + (half_ << 3);
      const int bb = grow >> 10, t = grow & (T_ - 1);
      const half_t val = (half_t)acc[r];
      if (which == 0)      qb[((bb * H_ + h) * T_ + t) * DK_ + sub] = val;
      else if (which == 1) kb[((bb * H_ + h) * T_ + t) * DK_ + sub] = val;
      else                 vtb[((bb * H_ + h) * DH_ + sub) * T_ + t] = val;
    }
  }
}

// ---------------------------------------------------------------------------
// Fused attention: one block per (b, h, 16-row tile).
// Phase 0: kick off async global->LDS copy of this head's V^T slab (64 KB),
//          overlapped behind phases 1-2; drained with s_wait_asynccnt.
// Phase 1: QK^T via WMMA, fused pos-bias MLP + mask into the epilogue,
//          full 16x1024 logit stripe in LDS.
// Phase 2: row softmax with all-masked (fix_mask) fallback -> f16 probs.
// Phase 3: probs x V via WMMA from LDS, partials reduced via LDS atomics.
// LDS ~164 KB -- only possible on CDNA5's 320 KB-per-WGP LDS.
// ---------------------------------------------------------------------------
__global__ __launch_bounds__(128) void tel_attention(
    const half_t* __restrict__ qb, const half_t* __restrict__ kb,
    const half_t* __restrict__ vtb,
    const float* __restrict__ rel, const unsigned char* __restrict__ mask,
    const float* __restrict__ pw1, const float* __restrict__ pb1,
    const float* __restrict__ pw2, const float* __restrict__ pb2,
    half_t* __restrict__ headsb) {
  __shared__ float  slog[16][T_ + 8];
  __shared__ half_t sprob[16][T_];
  __shared__ half_t sv[DH_][T_];
  __shared__ float  sacc[16][DH_];

  const int tid  = threadIdx.x;
  const int lane = tid & 31;
  const int w    = tid >> 5;
  const int bh = blockIdx.x >> 6;          // / (T/16)
  const int tt = blockIdx.x & 63;
  const int b = bh >> 3, h = bh & 7;
  const int t0 = tt << 4;

  // ---- Phase 0: async-stage V^T[bh] (32 x 1024 f16) into LDS ----
  {
    const half_t* src = vtb + (size_t)bh * DH_ * T_;
    const unsigned lds0 = (unsigned)(unsigned long long)(&sv[0][0]);
    for (int i = tid; i < (DH_ * T_) / 8; i += 128)   // 8 halves per b128
      async_ld_b128(lds0 + i * 16, src + i * 8);
  }

  for (int i = tid; i < 16 * DH_; i += 128) ((float*)sacc)[i] = 0.0f;

  float w1[PH_], b1r[PH_], w2[PH_];
#pragma unroll
  for (int p = 0; p < PH_; ++p) {
    w1[p]  = pw1[h * PH_ + p];
    b1r[p] = pb1[h * PH_ + p];
    w2[p]  = pw2[h * PH_ + p];
  }
  const float b2 = pb2[h];

  const v16h aq = load_a16(qb + (bh * T_ + t0) * DK_, DK_, lane);
  const int half_ = lane >> 4;
  const int n = lane & 15;

  // ---- Phase 1: logits ----
  for (int st = w; st < 64; st += 4) {
    const int s0 = st << 4;
    v16h bk = load_b16(kb + (bh * T_ + s0) * DK_, DK_, lane);
    v8f c = {};
    c = wmma16(aq, bk, c);
    const int s = s0 + n;
#pragma unroll
    for (int r = 0; r < 8; ++r) {
      const int m = r + (half_ << 3);
      const int t = t0 + m;
      const float rr = rel[(b * T_ + t) * T_ + s];
      float bias = b2;
#pragma unroll
      for (int p = 0; p < PH_; ++p) {
        float hid = rr * w1[p] + b1r[p];
        hid = hid > 0.0f ? hid : 0.01f * hid;   // leaky_relu(0.01)
        bias += hid * w2[p];
      }
      float lg = c[r] * INV_SQRT_DK - bias;
      if (mask[(b * T_ + t) * T_ + s]) lg = kNegBig;
      slog[m][s] = lg;
    }
  }
  __syncthreads();

  // ---- Phase 2: softmax (4 rows per wave) ----
  for (int m = w * 4; m < w * 4 + 4; ++m) {
    float mx = -3.4e38f;
    for (int s = lane; s < T_; s += 32) mx = fmaxf(mx, slog[m][s]);
#pragma unroll
    for (int off = 16; off; off >>= 1) mx = fmaxf(mx, __shfl_xor(mx, off, 32));
    const bool fix = mx < kNegThr;      // all-masked row -> logits := 1.0 -> uniform
    float sum = 0.0f;
    for (int s = lane; s < T_; s += 32) {
      float p = fix ? 1.0f : __expf(slog[m][s] - mx);
      slog[m][s] = p;
      sum += p;
    }
#pragma unroll
    for (int off = 16; off; off >>= 1) sum += __shfl_xor(sum, off, 32);
    const float rinv = 1.0f / sum;
    for (int s = lane; s < T_; s += 32)
      sprob[m][s] = (half_t)(slog[m][s] * rinv);
  }
  wait_async0();      // drain this wave's async V copies before consuming LDS
  __syncthreads();

  // ---- Phase 3: probs x V from LDS. waves (0,2)->cols 0..15, (1,3)->16..31 ----
  const int n0 = (w & 1) << 4;
  v8f acc = {};
  for (int kk = (w >> 1); kk < 32; kk += 2) {
    const int kc = kk << 5;
    v16h ap = load_a16(&sprob[0][0] + kc, T_, lane);
    v16h bv = load_b16(&sv[n0][0] + kc, T_, lane);
    acc = wmma16(ap, bv, acc);
  }
#pragma unroll
  for (int r = 0; r < 8; ++r) {
    const int m = r + (half_ << 3);
    atomicAdd(&sacc[m][n0 + n], acc[r]);
  }
  __syncthreads();

  for (int i = tid; i < 16 * DH_; i += 128) {
    const int m = i >> 5, nn = i & 31;
    headsb[(b * T_ + t0 + m) * D_ + h * DH_ + nn] = (half_t)sacc[m][nn];
  }
}

// ---------------------------------------------------------------------------
// Output projection: y = heads @ Wo + bo  (fp32 out), 16x32 tile per wave.
// ---------------------------------------------------------------------------
__global__ __launch_bounds__(32) void tel_wo_gemm(
    const half_t* __restrict__ hb, const half_t* __restrict__ Wot,
    const float* __restrict__ bo, float* __restrict__ y) {
  const int lane = threadIdx.x;
  const int m0 = blockIdx.x << 4;
  const int n0 = blockIdx.y << 5;
  v8f acc0 = {}, acc1 = {};
  for (int kc = 0; kc < D_; kc += 32) {
    if (kc + 32 < D_) __builtin_prefetch(hb + m0 * D_ + kc + 32, 0, 1);
    v16h a  = load_a16(hb + m0 * D_ + kc, D_, lane);
    v16h b0 = load_b16(Wot + n0 * D_ + kc, D_, lane);
    v16h b1 = load_b16(Wot + (n0 + 16) * D_ + kc, D_, lane);
    acc0 = wmma16(a, b0, acc0);
    acc1 = wmma16(a, b1, acc1);
  }
  const int half_ = lane >> 4;
#pragma unroll
  for (int nt = 0; nt < 2; ++nt) {
    const v8f& acc = nt ? acc1 : acc0;
    const int gcol = n0 + nt * 16 + (lane & 15);
    const float bias = bo[gcol];
#pragma unroll
    for (int r = 0; r < 8; ++r) {
      const int grow = m0 + r + (half_ << 3);
      y[grow * D_ + gcol] = acc[r] + bias;
    }
  }
}

// ---------------------------------------------------------------------------
// LayerNorm over D=256, one row per wave, optional leaky-relu on second input.
// mode 0: out = LN(a + b), also emit f16 copy. mode 1: out = LN(a + leaky(b)).
// ---------------------------------------------------------------------------
__global__ __launch_bounds__(128) void tel_layernorm(
    const float* __restrict__ a, const float* __restrict__ bsrc,
    const float* __restrict__ g, const float* __restrict__ be,
    float* __restrict__ out, half_t* __restrict__ outh, int leaky) {
  const int lane = threadIdx.x & 31;
  const int row = blockIdx.x * 4 + (threadIdx.x >> 5);
  const float* pa = a + row * D_;
  const float* pb = bsrc + row * D_;
  float v[8];
  float s = 0.0f;
#pragma unroll
  for (int j = 0; j < 8; ++j) {
    const int c = lane + j * 32;
    float t = pb[c];
    if (leaky) t = t > 0.0f ? t : 0.01f * t;
    v[j] = pa[c] + t;
    s += v[j];
  }
#pragma unroll
  for (int off = 16; off; off >>= 1) s += __shfl_xor(s, off, 32);
  const float mu = s * (1.0f / D_);
  float q = 0.0f;
#pragma unroll
  for (int j = 0; j < 8; ++j) { const float d = v[j] - mu; q += d * d; }
#pragma unroll
  for (int off = 16; off; off >>= 1) q += __shfl_xor(q, off, 32);
  const float rstd = rsqrtf(q * (1.0f / D_) + LN_EPS);
#pragma unroll
  for (int j = 0; j < 8; ++j) {
    const int c = lane + j * 32;
    const float o = (v[j] - mu) * rstd * g[c] + be[c];
    out[row * D_ + c] = o;
    if (outh) outh[row * D_ + c] = (half_t)o;
  }
}

// ---------------------------------------------------------------------------
// Causal conv (CS=3) as 3 accumulated WMMA GEMMs with row-shifted A tiles.
// Rows with t + shift < 0 (batch start) load zero fragments (padding).
// ---------------------------------------------------------------------------
__device__ __forceinline__ v16h load_a16_shift(const half_t* zb, int m0,
                                               int shift, int kc, int lane) {
  const int row = m0 + (lane & 15);
  const int t = row & (T_ - 1);
  v16h f;
#pragma unroll
  for (int e = 0; e < 16; ++e) f[e] = (half_t)0.0f;
  if (t + shift >= 0) {
    const half_t* p = zb + (row + shift) * D_ + kc + ((lane >> 4) << 3);
#pragma unroll
    for (int j = 0; j < 4; ++j) {
      f[2 * j]     = p[2 * j];
      f[2 * j + 1] = p[2 * j + 1];
      f[8 + 2 * j]     = p[16 + 2 * j];
      f[8 + 2 * j + 1] = p[16 + 2 * j + 1];
    }
  }
  return f;
}

__global__ __launch_bounds__(32) void tel_conv_gemm(
    const half_t* __restrict__ zb, const half_t* __restrict__ Wcb,
    const float* __restrict__ bc, float* __restrict__ zff) {
  const int lane = threadIdx.x;
  const int m0 = blockIdx.x << 4;
  const int n0 = blockIdx.y << 5;
  v8f acc0 = {}, acc1 = {};
#pragma unroll
  for (int c = 0; c < CS_; ++c) {
    const int shift = c - (CS_ - 1);
    const half_t* Wc_c = Wcb + c * D_ * D_;
    for (int kc = 0; kc < D_; kc += 32) {
      v16h a  = load_a16_shift(zb, m0, shift, kc, lane);
      v16h b0 = load_b16(Wc_c + n0 * D_ + kc, D_, lane);
      v16h b1 = load_b16(Wc_c + (n0 + 16) * D_ + kc, D_, lane);
      acc0 = wmma16(a, b0, acc0);
      acc1 = wmma16(a, b1, acc1);
    }
  }
  const int half_ = lane >> 4;
#pragma unroll
  for (int nt = 0; nt < 2; ++nt) {
    const v8f& acc = nt ? acc1 : acc0;
    const int gcol = n0 + nt * 16 + (lane & 15);
    const float bias = bc[gcol];
#pragma unroll
    for (int r = 0; r < 8; ++r) {
      const int grow = m0 + r + (half_ << 3);
      zff[grow * D_ + gcol] = acc[r] + bias;
    }
  }
}

// ---------------------------------------------------------------------------
// Launch
// ---------------------------------------------------------------------------
extern "C" void kernel_launch(void* const* d_in, const int* in_sizes, int n_in,
                              void* d_out, int out_size, void* d_ws, size_t ws_size,
                              hipStream_t stream) {
  (void)in_sizes; (void)n_in; (void)out_size; (void)ws_size;

  const float* x            = (const float*)d_in[0];
  const unsigned char* mask = (const unsigned char*)d_in[1];
  const float* rel          = (const float*)d_in[2];
  const float* Wq           = (const float*)d_in[3];
  const float* Wk           = (const float*)d_in[4];
  const float* Wv           = (const float*)d_in[5];
  const float* pw1          = (const float*)d_in[6];
  const float* pb1          = (const float*)d_in[7];
  const float* pw2          = (const float*)d_in[8];
  const float* pb2          = (const float*)d_in[9];
  const float* Wo           = (const float*)d_in[10];
  const float* bo           = (const float*)d_in[11];
  const float* Wc           = (const float*)d_in[12];
  const float* bc           = (const float*)d_in[13];
  const float* ln1g         = (const float*)d_in[14];
  const float* ln1b         = (const float*)d_in[15];
  const float* ln2g         = (const float*)d_in[16];
  const float* ln2b         = (const float*)d_in[17];
  float* out = (float*)d_out;

  char* ws = (char*)d_ws;
  size_t off = 0;
  auto alloc = [&](size_t bytes) {
    char* p = ws + off;
    off += (bytes + 255) & ~(size_t)255;
    return (void*)p;
  };
  half_t* xb  = (half_t*)alloc((size_t)BT_ * D_ * 2);        // x f16
  half_t* qb  = (half_t*)alloc((size_t)BT_ * D_ * 2);        // [B,H,T,32]
  half_t* kb  = (half_t*)alloc((size_t)BT_ * D_ * 2);        // [B,H,T,32]
  half_t* vtb = (half_t*)alloc((size_t)BT_ * D_ * 2);        // [B,H,32,T]
  half_t* hb  = (half_t*)alloc((size_t)BT_ * D_ * 2);        // heads [BT,256]
  half_t* zb  = (half_t*)alloc((size_t)BT_ * D_ * 2);        // z f16
  half_t* Wqt = (half_t*)alloc((size_t)D_ * D_ * 2);
  half_t* Wkt = (half_t*)alloc((size_t)D_ * D_ * 2);
  half_t* Wvt = (half_t*)alloc((size_t)D_ * D_ * 2);
  half_t* Wot = (half_t*)alloc((size_t)D_ * D_ * 2);
  half_t* Wcb = (half_t*)alloc((size_t)CS_ * D_ * D_ * 2);
  float* ybuf = (float*)alloc((size_t)BT_ * D_ * 4);
  float* zbuf = (float*)alloc((size_t)BT_ * D_ * 4);
  float* zff  = (float*)alloc((size_t)BT_ * D_ * 4);

  tel_cast_x<<<(BT_ * D_ + 255) / 256, 256, 0, stream>>>(x, xb, BT_ * D_);
  tel_prep_w<<<(D_ * D_) / 256, 256, 0, stream>>>(Wq, Wk, Wv, Wo, Wc,
                                                  Wqt, Wkt, Wvt, Wot, Wcb);
  tel_qkv_gemm<<<dim3(BT_ / 16, D_ / 32, 3), 32, 0, stream>>>(
      xb, Wqt, Wkt, Wvt, qb, kb, vtb);
  tel_attention<<<B_ * H_ * (T_ / 16), 128, 0, stream>>>(
      qb, kb, vtb, rel, mask, pw1, pb1, pw2, pb2, hb);
  tel_wo_gemm<<<dim3(BT_ / 16, D_ / 32), 32, 0, stream>>>(hb, Wot, bo, ybuf);
  tel_layernorm<<<BT_ / 4, 128, 0, stream>>>(x, ybuf, ln1g, ln1b, zbuf, zb, 0);
  tel_conv_gemm<<<dim3(BT_ / 16, D_ / 32), 32, 0, stream>>>(zb, Wcb, bc, zff);
  tel_layernorm<<<BT_ / 4, 128, 0, stream>>>(zbuf, zff, ln2g, ln2b, out,
                                             (half_t*)nullptr, 1);
}
